// BiMultiHeadAttention_11682311045282
// MI455X (gfx1250) — compile-verified, matrix operand order checked by reading, not run on hardware
//
#include <hip/hip_runtime.h>
#include <hip/hip_bf16.h>

typedef __bf16 bf16;
typedef __attribute__((ext_vector_type(16))) __bf16 v16bf;
typedef __attribute__((ext_vector_type(8)))  float  v8f;
typedef __attribute__((ext_vector_type(4)))  unsigned int u32x4;
typedef __attribute__((ext_vector_type(8)))  unsigned int u32x8;

#define B_DIM   2
#define T_DIM   16000
#define S_DIM   256
#define L_DIM   768
#define EMBED   256
#define HEADS   8
#define HD      32
#define SCALE_F 0.17677669529663687f  /* 32^-0.5 */
#define CLAMP_V 50000.0f
#define MASK_NEG -9e15f

union AF { v16bf v; bf16 e[16]; uint4 q[2]; };
union CF { v8f  v; float e[8]; };

// Load a 16x32 bf16 fragment (A layout; B operands load as A-of-transpose)
// from row-major storage: element[m][k] = p[(row0+m)*ld + k0 + k].
// Per ISA 7.12.2: lane<16 holds row m=lane, K {0..7,16..23}; lane>=16 same row,
// K {8..15,24..31}. Each lane: two contiguous 16B loads (global or ds).
__device__ inline v16bf frag_load(const bf16* __restrict__ p, int ld,
                                  int row0, int k0, int lane) {
  const bf16* base = p + (size_t)(row0 + (lane & 15)) * (size_t)ld
                       + k0 + ((lane >> 4) << 3);
  AF f;
  f.q[0] = *(const uint4*)(base);
  f.q[1] = *(const uint4*)(base + 16);
  return f.v;
}

__device__ inline void wmma_bf16(v16bf a, v16bf b, v8f* c) {
  *c = __builtin_amdgcn_wmma_f32_16x16x32_bf16(false, a, false, b,
                                               (short)0, *c, false, false);
}

// ---- Tensor Data Mover: 2-D tile (tile_x contiguous elems, tile_y rows of
// stride_elems) of 2-byte data from global -> LDS. D# per CDNA5 ISA 8.3/8.4:
// group0: count=1, lds_addr, global_addr, type=2; group1: data_size=1(2B),
// tensor_dim == tile_dim (no OOB), dim0 stride. VADDR2/3 = null (2-D tensor).
__device__ inline void tdm_load_2d_to_lds(void* lds_ptr, const void* gptr,
                                          unsigned tile_x, unsigned tile_y,
                                          unsigned long long stride_elems) {
  unsigned long long ga = (unsigned long long)gptr;
  unsigned lds_off = (unsigned)(unsigned long long)lds_ptr; // low 32b = LDS offset
  u32x4 g0;
  g0[0] = 1u;                                               // count=1, user mode
  g0[1] = lds_off;                                          // lds_addr
  g0[2] = (unsigned)ga;                                     // global_addr[31:0]
  g0[3] = (unsigned)((ga >> 32) & 0x01ffffffu) | 0x80000000u; // ga[56:32]|type=2
  u32x8 g1;
  g1[0] = 1u << 16;                                         // data_size=2 bytes
  g1[1] = (tile_x & 0xffffu) << 16;                         // tensor_dim0[15:0]
  g1[2] = ((tile_x >> 16) & 0xffffu) | ((tile_y & 0xffffu) << 16); // td0/td1
  g1[3] = ((tile_y >> 16) & 0xffffu) | ((tile_x & 0xffffu) << 16); // tile_dim0
  g1[4] = tile_y & 0xffffu;                                 // tile_dim1
  g1[5] = (unsigned)(stride_elems & 0xffffffffu);           // dim0_stride lo
  g1[6] = (unsigned)((stride_elems >> 32) & 0xffffu);       // dim0_stride hi
  g1[7] = 0u;
  asm volatile("tensor_load_to_lds %0, %1, null, null"
               :: "s"(g0), "s"(g1) : "memory");
}

__device__ inline float hr_max(float v) {   // reduce across 16-lane half
  v = fmaxf(v, __shfl_xor(v, 1, 32));
  v = fmaxf(v, __shfl_xor(v, 2, 32));
  v = fmaxf(v, __shfl_xor(v, 4, 32));
  v = fmaxf(v, __shfl_xor(v, 8, 32));
  return v;
}
__device__ inline float hr_sum(float v) {
  v += __shfl_xor(v, 1, 32);
  v += __shfl_xor(v, 2, 32);
  v += __shfl_xor(v, 4, 32);
  v += __shfl_xor(v, 8, 32);
  return v;
}

__device__ inline void atomicMaxF(float* a, float v) {
  if (!(v < 0.0f)) atomicMax((int*)a, __float_as_int(v));
  else             atomicMin((unsigned int*)a, (unsigned int)__float_as_int(v));
}

// ---------------------------------------------------------------- conversions
__global__ void cvt_bf16_kernel(const float* __restrict__ s,
                                bf16* __restrict__ d, int n) {
  int i = blockIdx.x * blockDim.x + threadIdx.x;
  if (i < n) d[i] = (bf16)s[i];
}

__global__ void init_stats_kernel(float* __restrict__ colmax,
                                  float* __restrict__ denom,
                                  float* __restrict__ numer,
                                  int nCM, int nNum) {
  int i = blockIdx.x * blockDim.x + threadIdx.x;
  if (i < nCM) { colmax[i] = -__builtin_inff(); denom[i] = 0.0f; }
  if (i < nNum) numer[i] = 0.0f;
}

// ------------------------------------------------------- generic bf16 WMMA GEMM
// Y[m][n] = (sum_k X[m][k]*W[n][k] + bias[n]) * scale
// 128 threads = 4 waves; wave -> 64x32 macro-tile (4 A-frags x 2 B-frags,
// 8 WMMAs/K-step); block covers 64 rows x 128 cols; grid (M/64, N/128).
__global__ __launch_bounds__(128)
void gemm_bf16_kernel(const bf16* __restrict__ X, int lda,
                      const bf16* __restrict__ W, int ldb,
                      const float* __restrict__ bias, float scale, int Kdim,
                      float* __restrict__ outF,
                      bf16*  __restrict__ outB, int ldo,
                      bf16*  __restrict__ outT, int Mtot) {
  int lane = threadIdx.x & 31, w = threadIdx.x >> 5;
  int m0 = blockIdx.x * 64;
  int n0 = blockIdx.y * 128 + w * 32;
  int lo = lane & 15, hi = lane >> 4;

  CF acc[4][2];
  for (int im = 0; im < 4; im++)
    for (int in = 0; in < 2; in++)
      for (int i = 0; i < 8; i++) acc[im][in].e[i] = 0.0f;

  for (int k0 = 0; k0 < Kdim; k0 += 32) {
    v16bf a[4], bb[2];
#pragma unroll
    for (int im = 0; im < 4; im++) a[im] = frag_load(X, lda, m0 + 16 * im, k0, lane);
#pragma unroll
    for (int in = 0; in < 2; in++) bb[in] = frag_load(W, ldb, n0 + 16 * in, k0, lane);
#pragma unroll
    for (int im = 0; im < 4; im++)
#pragma unroll
      for (int in = 0; in < 2; in++)
        wmma_bf16(a[im], bb[in], &acc[im][in].v);
  }

  for (int im = 0; im < 4; im++) {
    for (int in = 0; in < 2; in++) {
      int mrow = m0 + 16 * im;
      int ncol = n0 + 16 * in + lo;
      float bia = bias ? bias[ncol] : 0.0f;
      float r[8];
      for (int i = 0; i < 8; i++) r[i] = (acc[im][in].e[i] + bia) * scale;
      if (outF) {
        for (int i = 0; i < 8; i++)
          outF[(size_t)(mrow + i + 8 * hi) * (size_t)ldo + ncol] = r[i];
      }
      if (outB) {
        for (int i = 0; i < 8; i++)
          outB[(size_t)(mrow + i + 8 * hi) * (size_t)ldo + ncol] = (bf16)r[i];
      }
      if (outT) { // transposed (N x Mtot): 8 consecutive rows -> one 16B store
        bf16 t8[8];
        for (int i = 0; i < 8; i++) t8[i] = (bf16)r[i];
        *(uint4*)(outT + (size_t)ncol * (size_t)Mtot + mrow + 8 * hi) =
            *(const uint4*)t8;
      }
    }
  }
}

// -------------------------------------------- pass 1: vision softmax + out_v
// grid (T/128, B*H), 256 threads = 8 waves; wave w owns rows t0..t0+15, all S.
// K-head tile (256x32) and Vl-head tile (32x256) staged once per block via TDM.
__global__ __launch_bounds__(256)
void attn_v_kernel(const bf16* __restrict__ Qb, const bf16* __restrict__ Kb,
                   const bf16* __restrict__ VlT, const int* __restrict__ maskl,
                   float* __restrict__ colmax, bf16* __restrict__ Ovb) {
  __shared__ __align__(16) bf16 Ksh[S_DIM * HD];     // (s, d)  ld=32
  __shared__ __align__(16) bf16 Vsh[HD * S_DIM];     // (d, s)  ld=256
  __shared__ __align__(16) bf16 P2[8][16][32];
  int lane = threadIdx.x & 31, w = threadIdx.x >> 5;
  int lo = lane & 15, hi = lane >> 4;
  int bh = blockIdx.y, b = bh >> 3, h = bh & 7;
  int t0 = blockIdx.x * 128 + w * 16;

  if (w == 0) {
    tdm_load_2d_to_lds(Ksh, Kb + (size_t)(b * S_DIM) * EMBED + h * HD,
                       HD, S_DIM, EMBED);
    tdm_load_2d_to_lds(Vsh, VlT + (size_t)(h * HD) * (B_DIM * S_DIM) + b * S_DIM,
                       S_DIM, HD, B_DIM * S_DIM);
    __builtin_amdgcn_s_wait_tensorcnt(0);
  }
  __syncthreads();

  v16bf aq = frag_load(Qb, EMBED, b * T_DIM + t0, h * HD, lane);

  CF c[16];
  for (int j = 0; j < 16; j++) {
    v16bf bk = frag_load(Ksh, HD, j * 16, 0, lane);
    CF t; for (int i = 0; i < 8; i++) t.e[i] = 0.0f;
    wmma_bf16(aq, bk, &t.v);
    float lm = -__builtin_inff();
    for (int i = 0; i < 8; i++) {
      t.e[i] = fminf(fmaxf(t.e[i], -CLAMP_V), CLAMP_V);
      lm = fmaxf(lm, t.e[i]);
    }
    lm = fmaxf(lm, __shfl_xor(lm, 16, 32));        // fold the two half-waves
    if (hi == 0) atomicMaxF(&colmax[bh * S_DIM + j * 16 + lo], lm);
    float mv = (maskl[b * S_DIM + j * 16 + lo] == 0) ? MASK_NEG : 0.0f;
    for (int i = 0; i < 8; i++) t.e[i] += mv;
    c[j] = t;
  }

  float rm[8], rs[8];
  for (int i = 0; i < 8; i++) rm[i] = -__builtin_inff();
  for (int j = 0; j < 16; j++)
    for (int i = 0; i < 8; i++) rm[i] = fmaxf(rm[i], c[j].e[i]);
  for (int i = 0; i < 8; i++) rm[i] = hr_max(rm[i]);
  for (int i = 0; i < 8; i++) rs[i] = 0.0f;
  for (int j = 0; j < 16; j++)
    for (int i = 0; i < 8; i++) {
      float p = __expf(c[j].e[i] - rm[i]);
      c[j].e[i] = p; rs[i] += p;
    }
  for (int i = 0; i < 8; i++) rs[i] = hr_sum(rs[i]);

  CF o[2];
  for (int n = 0; n < 2; n++)
    for (int i = 0; i < 8; i++) o[n].e[i] = 0.0f;
  for (int ks = 0; ks < 8; ks++) {
    for (int jj = 0; jj < 2; jj++) {               // stage P chunk (C->A layout)
      int j = 2 * ks + jj;
      for (int i = 0; i < 8; i++)
        P2[w][i + 8 * hi][jj * 16 + lo] = (bf16)c[j].e[i];
    }
    v16bf ap = frag_load(&P2[w][0][0], 32, 0, 0, lane);
    for (int n = 0; n < 2; n++) {
      v16bf bv = frag_load(Vsh, S_DIM, n * 16, ks * 32, lane);
      wmma_bf16(ap, bv, &o[n].v);
    }
  }
  for (int n = 0; n < 2; n++)
    for (int i = 0; i < 8; i++)
      Ovb[(size_t)(b * T_DIM + t0 + i + 8 * hi) * EMBED + h * HD + n * 16 + lo] =
          (bf16)(o[n].e[i] / rs[i]);
}

// ----------------------------- pass 2: language softmax partials (numer/denom)
// grid (T/128, B*H), 256 threads = 8 waves; wave w owns 2 s-row-tiles, 128 t-cols.
// Q chunk (128x32) and Vv chunk (32x128) staged once per block via TDM.
__global__ __launch_bounds__(256)
void attn_l_kernel(const bf16* __restrict__ Qb, const bf16* __restrict__ Kb,
                   const bf16* __restrict__ VvT, const float* __restrict__ colmax,
                   float* __restrict__ numer, float* __restrict__ denom) {
  __shared__ __align__(16) bf16 Qsh[128 * HD];       // (t_local, d) ld=32
  __shared__ __align__(16) bf16 Vsh[HD * 128];       // (d, t_local) ld=128
  __shared__ __align__(16) bf16 P2[8][16][32];
  int lane = threadIdx.x & 31, w = threadIdx.x >> 5;
  int lo = lane & 15, hi = lane >> 4;
  int bh = blockIdx.y, b = bh >> 3, h = bh & 7;
  int tbase = blockIdx.x * 128;

  if (w == 0) {
    tdm_load_2d_to_lds(Qsh, Qb + (size_t)(b * T_DIM + tbase) * EMBED + h * HD,
                       HD, 128, EMBED);
    tdm_load_2d_to_lds(Vsh,
                       VvT + (size_t)(h * HD) * (B_DIM * T_DIM) + (size_t)b * T_DIM + tbase,
                       128, HD, B_DIM * T_DIM);
    __builtin_amdgcn_s_wait_tensorcnt(0);
  }
  __syncthreads();

  for (int p = 0; p < 2; p++) {
    int s0 = (w * 2 + p) * 16;
    v16bf ak = frag_load(Kb, EMBED, b * S_DIM + s0, h * HD, lane);
    float cm[8];
    for (int i = 0; i < 8; i++) cm[i] = colmax[bh * S_DIM + s0 + i + 8 * hi];

    CF acc[2]; float dsum[8];
    for (int n = 0; n < 2; n++)
      for (int i = 0; i < 8; i++) acc[n].e[i] = 0.0f;
    for (int i = 0; i < 8; i++) dsum[i] = 0.0f;

    for (int jc = 0; jc < 4; jc++) {
      for (int jj = 0; jj < 2; jj++) {
        v16bf bq = frag_load(Qsh, HD, jc * 32 + jj * 16, 0, lane);
        CF t; for (int i = 0; i < 8; i++) t.e[i] = 0.0f;
        wmma_bf16(ak, bq, &t.v);
        for (int i = 0; i < 8; i++) {
          float a = fminf(fmaxf(t.e[i], -CLAMP_V), CLAMP_V);
          float x = fmaxf(a - cm[i], -CLAMP_V);
          float pp = __expf(x);
          dsum[i] += pp;
          P2[w][i + 8 * hi][jj * 16 + lo] = (bf16)pp;
        }
      }
      v16bf ap = frag_load(&P2[w][0][0], 32, 0, 0, lane);
      for (int n = 0; n < 2; n++) {
        v16bf bv = frag_load(Vsh, 128, n * 16, jc * 32, lane);
        wmma_bf16(ap, bv, &acc[n].v);
      }
    }
    for (int n = 0; n < 2; n++)
      for (int i = 0; i < 8; i++)
        atomicAdd(&numer[((size_t)bh * S_DIM + s0 + i + 8 * hi) * HD + n * 16 + lo],
                  acc[n].e[i]);
    for (int i = 0; i < 8; i++) {
      float v = hr_sum(dsum[i]);
      if (lo == 0) atomicAdd(&denom[bh * S_DIM + s0 + i + 8 * hi], v);
    }
  }
}

__global__ void finalize_ol_kernel(const float* __restrict__ numer,
                                   const float* __restrict__ denom,
                                   bf16* __restrict__ Olb) {
  int i = blockIdx.x * blockDim.x + threadIdx.x;
  if (i >= B_DIM * HEADS * S_DIM * HD) return;
  int d = i & 31, s = (i >> 5) & 255, bh = i >> 13;
  int b = bh >> 3, h = bh & 7;
  Olb[(size_t)(b * S_DIM + s) * EMBED + h * HD + d] =
      (bf16)(numer[i] / denom[bh * S_DIM + s]);
}

// --------------------------------------------------------------------- launch
extern "C" void kernel_launch(void* const* d_in, const int* in_sizes, int n_in,
                              void* d_out, int out_size, void* d_ws, size_t ws_size,
                              hipStream_t stream) {
  const float* v    = (const float*)d_in[0];
  const float* l    = (const float*)d_in[1];
  const int*   mask = (const int*)d_in[2];
  const float* W_v  = (const float*)d_in[3];  const float* b_v  = (const float*)d_in[4];
  const float* W_l  = (const float*)d_in[5];  const float* b_l  = (const float*)d_in[6];
  const float* W_vv = (const float*)d_in[7];  const float* b_vv = (const float*)d_in[8];
  const float* W_vl = (const float*)d_in[9];  const float* b_vl = (const float*)d_in[10];
  const float* W_ov = (const float*)d_in[11]; const float* b_ov = (const float*)d_in[12];
  const float* W_ol = (const float*)d_in[13]; const float* b_ol = (const float*)d_in[14];

  char* ws = (char*)d_ws;
  size_t off = 0;
  auto alloc = [&](size_t bytes) {
    char* p = ws + off;
    off = (off + bytes + 255) & ~(size_t)255;
    return p;
  };
  const int NV = B_DIM * T_DIM;                 // 32000
  const int NL = B_DIM * S_DIM;                 // 512

  bf16* vB   = (bf16*)alloc((size_t)NV * EMBED * 2);
  bf16* lB   = (bf16*)alloc((size_t)NL * L_DIM * 2);
  bf16* WvB  = (bf16*)alloc((size_t)EMBED * EMBED * 2);
  bf16* WlB  = (bf16*)alloc((size_t)EMBED * L_DIM * 2);
  bf16* WvvB = (bf16*)alloc((size_t)EMBED * EMBED * 2);
  bf16* WvlB = (bf16*)alloc((size_t)EMBED * L_DIM * 2);
  bf16* WovB = (bf16*)alloc((size_t)EMBED * EMBED * 2);
  bf16* WolB = (bf16*)alloc((size_t)L_DIM * EMBED * 2);
  bf16* Qb   = (bf16*)alloc((size_t)NV * EMBED * 2);
  bf16* Kb   = (bf16*)alloc((size_t)NL * EMBED * 2);
  bf16* VvT  = (bf16*)alloc((size_t)EMBED * NV * 2);   // (256, B*T)
  bf16* VlT  = (bf16*)alloc((size_t)EMBED * NL * 2);   // (256, B*S)
  bf16* Ovb  = (bf16*)alloc((size_t)NV * EMBED * 2);
  bf16* Olb  = (bf16*)alloc((size_t)NL * EMBED * 2);
  float* colmax = (float*)alloc((size_t)B_DIM * HEADS * S_DIM * 4);
  float* denom  = (float*)alloc((size_t)B_DIM * HEADS * S_DIM * 4);
  float* numer  = (float*)alloc((size_t)B_DIM * HEADS * S_DIM * HD * 4);
  (void)ws_size; (void)n_in; (void)in_sizes; (void)out_size;

  auto cvt = [&](const float* s, bf16* d, int n) {
    cvt_bf16_kernel<<<(n + 255) / 256, 256, 0, stream>>>(s, d, n);
  };
  cvt(v, vB, NV * EMBED);
  cvt(l, lB, NL * L_DIM);
  cvt(W_v, WvB, EMBED * EMBED);
  cvt(W_l, WlB, EMBED * L_DIM);
  cvt(W_vv, WvvB, EMBED * EMBED);
  cvt(W_vl, WvlB, EMBED * L_DIM);
  cvt(W_ov, WovB, EMBED * EMBED);
  cvt(W_ol, WolB, L_DIM * EMBED);

  init_stats_kernel<<<(B_DIM * HEADS * S_DIM * HD + 255) / 256, 256, 0, stream>>>(
      colmax, denom, numer, B_DIM * HEADS * S_DIM, B_DIM * HEADS * S_DIM * HD);

  // Q = (v @ W_v^T + b_v) * SCALE        (bf16, row-major)
  gemm_bf16_kernel<<<dim3(NV / 64, EMBED / 128), 128, 0, stream>>>(
      vB, EMBED, WvB, EMBED, b_v, SCALE_F, EMBED,
      nullptr, Qb, EMBED, nullptr, 0);
  // Vv^T = (v @ W_vv^T + b_vv)^T         (bf16, (256, B*T))
  gemm_bf16_kernel<<<dim3(NV / 64, EMBED / 128), 128, 0, stream>>>(
      vB, EMBED, WvvB, EMBED, b_vv, 1.0f, EMBED,
      nullptr, nullptr, 0, VvT, NV);
  // K = l @ W_l^T + b_l                  (bf16, row-major)
  gemm_bf16_kernel<<<dim3(NL / 64, EMBED / 128), 128, 0, stream>>>(
      lB, L_DIM, WlB, L_DIM, b_l, 1.0f, L_DIM,
      nullptr, Kb, EMBED, nullptr, 0);
  // Vl^T = (l @ W_vl^T + b_vl)^T         (bf16, (256, B*S))
  gemm_bf16_kernel<<<dim3(NL / 64, EMBED / 128), 128, 0, stream>>>(
      lB, L_DIM, WvlB, L_DIM, b_vl, 1.0f, L_DIM,
      nullptr, nullptr, 0, VlT, NL);

  attn_v_kernel<<<dim3(T_DIM / 128, B_DIM * HEADS), 256, 0, stream>>>(
      Qb, Kb, VlT, mask, colmax, Ovb);
  attn_l_kernel<<<dim3(T_DIM / 128, B_DIM * HEADS), 256, 0, stream>>>(
      Qb, Kb, VvT, colmax, numer, denom);
  finalize_ol_kernel<<<(B_DIM * HEADS * S_DIM * HD + 255) / 256, 256, 0, stream>>>(
      numer, denom, Olb);

  float* out_v = (float*)d_out;                        // (B,T,256)
  float* out_l = out_v + (size_t)NV * EMBED;           // (B,S,768)
  gemm_bf16_kernel<<<dim3(NV / 64, EMBED / 128), 128, 0, stream>>>(
      Ovb, EMBED, WovB, EMBED, b_ov, 1.0f, EMBED,
      out_v, nullptr, EMBED, nullptr, 0);
  gemm_bf16_kernel<<<dim3(NL / 64, L_DIM / 128), 128, 0, stream>>>(
      Olb, EMBED, WolB, EMBED, b_ol, 1.0f, EMBED,
      out_l, nullptr, L_DIM, nullptr, 0);
}
